// CausalSelfAttention_67267777790080
// MI455X (gfx1250) — matrix operand order repaired and checked by reading, not compiled, not run on hardware
//
#include <hip/hip_runtime.h>

// Causal self-attention for MI455X (gfx1250, wave32, WMMA).
// Pipeline: cast->bf16, QKV GEMM (WMMA), RoPE+relayout, flash attention (WMMA),
// proj GEMM (WMMA, f32 out).  Workspace requirement: ~72 MB.
// LDS staging uses gfx1250 GLOBAL_LOAD_ASYNC_TO_LDS_B128 (ASYNCcnt-tracked,
// bypasses VGPRs) instead of load+wait+ds_store.

#define B_ 2
#define T_ 2048
#define C_ 1024
#define H_ 16
#define D_ 64

typedef __bf16 v16bf __attribute__((ext_vector_type(16)));
typedef __bf16 v8bf  __attribute__((ext_vector_type(8)));
typedef float  v8f   __attribute__((ext_vector_type(8)));

__device__ __forceinline__ __bf16 f2bf(float f) {
  unsigned u = __builtin_bit_cast(unsigned, f);
  unsigned r = (u + 0x7FFFu + ((u >> 16) & 1u)) >> 16;  // RNE
  unsigned short s = (unsigned short)r;
  return __builtin_bit_cast(__bf16, s);
}
__device__ __forceinline__ float bf2f(__bf16 h) {
  unsigned short s = __builtin_bit_cast(unsigned short, h);
  unsigned u = ((unsigned)s) << 16;
  return __builtin_bit_cast(float, u);
}
// Build a 16-element bf16 fragment from two (possibly non-adjacent) 16B chunks.
__device__ __forceinline__ v16bf load16(const __bf16* p0, const __bf16* p1) {
  union U { v16bf v; struct { v8bf lo, hi; } s; } u;
  u.s.lo = *(const v8bf*)p0;
  u.s.hi = *(const v8bf*)p1;
  return u.v;
}
__device__ __forceinline__ v8f zero8() {
  v8f z;
#pragma unroll
  for (int i = 0; i < 8; ++i) z[i] = 0.0f;
  return z;
}

// 16-byte async global->LDS copy (gfx1250). LDS byte address = low 32 bits of
// the flat address of a __shared__ object (ISA aperture rule: addr[31:0]).
// Tracked by ASYNCcnt, not LOADcnt; no VGPR data round trip.
__device__ __forceinline__ void async_copy16(void* lds_dst, const void* gsrc) {
  unsigned lds = (unsigned)(size_t)lds_dst;
  unsigned long long g = (unsigned long long)(size_t)gsrc;
  asm volatile("global_load_async_to_lds_b128 %0, %1, off"
               :: "v"(lds), "v"(g) : "memory");
}
__device__ __forceinline__ void wait_async() {
  asm volatile("s_wait_asynccnt 0x0" ::: "memory");
}

// ---------------------------------------------------------------------------
// f32 -> bf16 cast
// ---------------------------------------------------------------------------
__global__ __launch_bounds__(256) void cast_f32_bf16(const float* __restrict__ in,
                                                     __bf16* __restrict__ out, int n) {
  int i = blockIdx.x * blockDim.x + threadIdx.x;
  if (i < n) out[i] = f2bf(in[i]);
}

// ---------------------------------------------------------------------------
// C[M,N] = A[M,K] * W[N,K]^T  (both row-major bf16), WMMA f32 accumulate.
// Block: 128 threads = 4 waves, 64x128 output tile, K-step 32.
// Each wave computes 16x128 (1 A-fragment reused across 8 B-fragments =
// 8 WMMAs per K-step) to maximize flops per LDS byte.
// ---------------------------------------------------------------------------
template <bool OUT_BF16>
__global__ __launch_bounds__(128) void wmma_gemm_nt(const __bf16* __restrict__ A,
                                                    const __bf16* __restrict__ W,
                                                    void* __restrict__ Out,
                                                    int M, int N, int K) {
  __shared__ __bf16 As[64 * 32];    // 4 KB
  __shared__ __bf16 Bs[128 * 32];   // 8 KB
  const int tid = threadIdx.x;
  const int lane = tid & 31;
  const int wave = tid >> 5;
  const int mBase = blockIdx.x * 64;
  const int nBase = blockIdx.y * 128;
  (void)M;

  v8f acc[8];
#pragma unroll
  for (int nt = 0; nt < 8; ++nt) acc[nt] = zero8();

  for (int kb = 0; kb < K; kb += 32) {
    // Stage A (64x32): 256 chunks of 8 bf16, 2 per thread (async -> LDS)
#pragma unroll
    for (int c = 0; c < 2; ++c) {
      int idx = tid + c * 128;
      int row = idx >> 2;
      int off = (idx & 3) * 8;
      async_copy16(&As[row * 32 + off], &A[(size_t)(mBase + row) * K + kb + off]);
      if (kb + 32 < K)
        __builtin_prefetch(&A[(size_t)(mBase + row) * K + kb + 32 + off], 0, 1);
    }
    // Stage W (128x32): 512 chunks, 4 per thread (async -> LDS)
#pragma unroll
    for (int c = 0; c < 4; ++c) {
      int idx = tid + c * 128;
      int row = idx >> 2;
      int off = (idx & 3) * 8;
      async_copy16(&Bs[row * 32 + off], &W[(size_t)(nBase + row) * K + kb + off]);
      if (kb + 32 < K)
        __builtin_prefetch(&W[(size_t)(nBase + row) * K + kb + 32 + off], 0, 1);
    }
    wait_async();
    __syncthreads();

    // A fragment (16x32): lanes 0-15 hold K 0..7 & 16..23; lanes 16-31 hold 8..15 & 24..31
    const int mA = wave * 16 + (lane & 15);
    const int ka = (lane >> 4) * 8;
    v16bf af = load16(&As[mA * 32 + ka], &As[mA * 32 + ka + 16]);

    // B fragments (32x16): lane = column n; lanes 0-15 hold K 0..7 & 16..23; lanes 16-31 hold 8..15 & 24..31
    const int nB = lane & 15;
    const int ks = (lane >> 4) * 8;
#pragma unroll
    for (int nt = 0; nt < 8; ++nt) {
      v16bf bfr = load16(&Bs[(nt * 16 + nB) * 32 + ks],
                         &Bs[(nt * 16 + nB) * 32 + ks + 16]);
      acc[nt] = __builtin_amdgcn_wmma_f32_16x16x32_bf16(false, af, false, bfr,
                                                        (short)0, acc[nt], false, false);
    }
    __syncthreads();
  }

  // C/D layout: lane = N, rows striped over 8 VGPRs per lane-half
  const int n0 = lane & 15;
  const int mOut = mBase + wave * 16 + (lane >> 4) * 8;
#pragma unroll
  for (int i = 0; i < 8; ++i) {
    size_t o = (size_t)(mOut + i) * N + nBase + n0;
#pragma unroll
    for (int nt = 0; nt < 8; ++nt) {
      if constexpr (OUT_BF16) {
        ((__bf16*)Out)[o + nt * 16] = f2bf(acc[nt][i]);
      } else {
        ((float*)Out)[o + nt * 16] = acc[nt][i];
      }
    }
  }
}

// ---------------------------------------------------------------------------
// RoPE + split qkv[B*T,3C] into Q,K,V in [B,H,T,D] (bf16). 1/sqrt(D) folded
// into Q. One thread per (b,h,t, pair j) with pair (j, j+32).
// ---------------------------------------------------------------------------
__global__ __launch_bounds__(256) void rope_split(const __bf16* __restrict__ qkv,
                                                  __bf16* __restrict__ Q,
                                                  __bf16* __restrict__ Kh,
                                                  __bf16* __restrict__ Vh) {
  int idx = blockIdx.x * blockDim.x + threadIdx.x;  // B*H*T*32 = 2^21
  if (idx >= B_ * H_ * T_ * 32) return;
  int j = idx & 31;
  int t = (idx >> 5) & (T_ - 1);
  int h = (idx >> 16) & (H_ - 1);
  int b = idx >> 20;

  size_t rowOff = ((size_t)(b * T_ + t)) * (3 * C_);
  int col = h * D_ + j;
  float q0 = bf2f(qkv[rowOff + col]);
  float q1 = bf2f(qkv[rowOff + col + 32]);
  float k0 = bf2f(qkv[rowOff + C_ + col]);
  float k1 = bf2f(qkv[rowOff + C_ + col + 32]);
  float v0 = bf2f(qkv[rowOff + 2 * C_ + col]);
  float v1 = bf2f(qkv[rowOff + 2 * C_ + col + 32]);

  float invf = __powf(10000.0f, -(float)j * (1.0f / 32.0f));
  float ang = (float)t * invf;
  float sn = __sinf(ang), cs = __cosf(ang);
  const float qscale = 0.125f;  // 1/sqrt(64)

  size_t o = (((size_t)(b * H_ + h) * T_) + t) * D_ + j;
  Q[o]      = f2bf((q0 * cs - q1 * sn) * qscale);
  Q[o + 32] = f2bf((q1 * cs + q0 * sn) * qscale);
  Kh[o]      = f2bf(k0 * cs - k1 * sn);
  Kh[o + 32] = f2bf(k1 * cs + k0 * sn);
  Vh[o]      = f2bf(v0);
  Vh[o + 32] = f2bf(v1);
}

// ---------------------------------------------------------------------------
// Flash attention: block = 128 threads (4 waves), each block does a 64-query
// tile of one (b,h); each wave owns 16 query rows. Key loop step 32 with
// K staged async into LDS and V staged transposed. Online softmax in
// registers; the causal mask is applied only on the <=2 diagonal tiles.
// ---------------------------------------------------------------------------
__global__ __launch_bounds__(128) void flash_attn_kernel(const __bf16* __restrict__ Q,
                                                         const __bf16* __restrict__ Kh,
                                                         const __bf16* __restrict__ Vh,
                                                         __bf16* __restrict__ Y) {
  __shared__ __bf16 Ks[32 * 72];       // [key][d], padded stride
  __shared__ __bf16 Vt[64 * 40];       // [d][key], padded stride
  __shared__ __bf16 Ps[4 * 16 * 32];   // per-wave P staging (C-layout -> A-layout)

  const int tid = threadIdx.x, lane = tid & 31, wave = tid >> 5;
  const int qBase = blockIdx.x * 64;
  const int bh = blockIdx.y;
  const int b = bh >> 4, h = bh & 15;
  const size_t headOff = (size_t)bh * T_ * D_;

  const int ka = (lane >> 4) * 8;
  const int nB = lane & 15;
  const int ds16 = (lane >> 4) * 8;

  // Persistent Q fragments (A-layout) for this wave's 16 rows, d 0..31 / 32..63
  const int qRowFrag = qBase + wave * 16 + nB;
  const __bf16* qp = Q + headOff + (size_t)qRowFrag * D_;
  v16bf qf0 = load16(qp + ka, qp + ka + 16);
  v16bf qf1 = load16(qp + 32 + ka, qp + 32 + ka + 16);

  v8f acc0 = zero8(), acc1 = zero8(), acc2 = zero8(), acc3 = zero8();
  float mrow[8], lrow[8];
#pragma unroll
  for (int i = 0; i < 8; ++i) { mrow[i] = -1e30f; lrow[i] = 0.0f; }

  const int myQ0 = qBase + wave * 16 + (lane >> 4) * 8;
  const int kEnd = qBase + 64;  // causal bound for this block

  for (int kb = 0; kb < kEnd; kb += 32) {
    __syncthreads();
    // Stage K tile 32x64 async into LDS (row-major copy)
#pragma unroll
    for (int c = 0; c < 2; ++c) {
      int idx = tid + c * 128;
      int r = idx >> 3, off = (idx & 7) * 8;
      async_copy16(&Ks[r * 72 + off], &Kh[headOff + (size_t)(kb + r) * D_ + off]);
      if (kb + 32 < kEnd)
        __builtin_prefetch(&Kh[headOff + (size_t)(kb + 32 + r) * D_ + off], 0, 1);
    }
    // Stage V tile transposed through VGPRs: Vt[d][key]
    {
      int r = tid >> 2, ch = tid & 3;
      const __bf16* vp = Vh + headOff + (size_t)(kb + r) * D_ + ch * 16;
#pragma unroll
      for (int j = 0; j < 16; ++j) Vt[(ch * 16 + j) * 40 + r] = vp[j];
      if (kb + 32 < kEnd)
        __builtin_prefetch(&Vh[headOff + (size_t)(kb + 32 + r) * D_ + ch * 16], 0, 1);
    }
    wait_async();
    __syncthreads();

    // S = Q * K^T for two 16-key subtiles (reduce over d in two K=32 WMMAs each)
    v8f s0 = zero8(), s1 = zero8();
    {
      v16bf k00 = load16(&Ks[nB * 72 + ds16],        &Ks[nB * 72 + ds16 + 16]);
      v16bf k01 = load16(&Ks[nB * 72 + 32 + ds16],   &Ks[nB * 72 + 32 + ds16 + 16]);
      v16bf k10 = load16(&Ks[(16 + nB) * 72 + ds16], &Ks[(16 + nB) * 72 + ds16 + 16]);
      v16bf k11 = load16(&Ks[(16 + nB) * 72 + 32 + ds16],
                         &Ks[(16 + nB) * 72 + 32 + ds16 + 16]);
      s0 = __builtin_amdgcn_wmma_f32_16x16x32_bf16(false, qf0, false, k00, (short)0, s0, false, false);
      s0 = __builtin_amdgcn_wmma_f32_16x16x32_bf16(false, qf1, false, k01, (short)0, s0, false, false);
      s1 = __builtin_amdgcn_wmma_f32_16x16x32_bf16(false, qf0, false, k10, (short)0, s1, false, false);
      s1 = __builtin_amdgcn_wmma_f32_16x16x32_bf16(false, qf1, false, k11, (short)0, s1, false, false);
    }

    // Causal mask only on the (at most 2) diagonal tiles; uniform branch so
    // interior tiles skip all cndmask work.
    if (kb + 32 > qBase) {
#pragma unroll
      for (int i = 0; i < 8; ++i) {
        const int qi = myQ0 + i;
        if (kb + nB > qi)      s0[i] = -1e30f;
        if (kb + 16 + nB > qi) s1[i] = -1e30f;
      }
    }

    // Online softmax (row stats per accumulator VGPR, reductions across the
    // 16-lane half that owns each row)
    float p0[8], p1[8];
#pragma unroll
    for (int i = 0; i < 8; ++i) {
      float rm = fmaxf(s0[i], s1[i]);
      rm = fmaxf(rm, __shfl_xor(rm, 1, 32));
      rm = fmaxf(rm, __shfl_xor(rm, 2, 32));
      rm = fmaxf(rm, __shfl_xor(rm, 4, 32));
      rm = fmaxf(rm, __shfl_xor(rm, 8, 32));
      const float mNew = fmaxf(mrow[i], rm);
      const float scale = __expf(mrow[i] - mNew);
      mrow[i] = mNew;
      const float e0 = __expf(s0[i] - mNew);
      const float e1 = __expf(s1[i] - mNew);
      float rs = e0 + e1;
      rs += __shfl_xor(rs, 1, 32);
      rs += __shfl_xor(rs, 2, 32);
      rs += __shfl_xor(rs, 4, 32);
      rs += __shfl_xor(rs, 8, 32);
      lrow[i] = lrow[i] * scale + rs;
      p0[i] = e0; p1[i] = e1;
      acc0[i] *= scale; acc1[i] *= scale; acc2[i] *= scale; acc3[i] *= scale;
    }

    // Re-layout P (C-format) -> A-fragment via per-wave LDS
    __bf16* Pw = &Ps[wave * 512];
#pragma unroll
    for (int i = 0; i < 8; ++i) {
      const int m = (lane >> 4) * 8 + i;
      Pw[m * 32 + nB]      = f2bf(p0[i]);
      Pw[m * 32 + 16 + nB] = f2bf(p1[i]);
    }
    asm volatile("s_wait_dscnt 0" ::: "memory");  // intra-wave LDS RAW fence
    const int mA = lane & 15;
    v16bf pf = load16(&Pw[mA * 32 + ka], &Pw[mA * 32 + ka + 16]);

    // O += P * V (four d-tiles of 16)
    v16bf vf0 = load16(&Vt[(0 * 16 + nB) * 40 + ds16], &Vt[(0 * 16 + nB) * 40 + ds16 + 16]);
    v16bf vf1 = load16(&Vt[(1 * 16 + nB) * 40 + ds16], &Vt[(1 * 16 + nB) * 40 + ds16 + 16]);
    v16bf vf2 = load16(&Vt[(2 * 16 + nB) * 40 + ds16], &Vt[(2 * 16 + nB) * 40 + ds16 + 16]);
    v16bf vf3 = load16(&Vt[(3 * 16 + nB) * 40 + ds16], &Vt[(3 * 16 + nB) * 40 + ds16 + 16]);
    acc0 = __builtin_amdgcn_wmma_f32_16x16x32_bf16(false, pf, false, vf0, (short)0, acc0, false, false);
    acc1 = __builtin_amdgcn_wmma_f32_16x16x32_bf16(false, pf, false, vf1, (short)0, acc1, false, false);
    acc2 = __builtin_amdgcn_wmma_f32_16x16x32_bf16(false, pf, false, vf2, (short)0, acc2, false, false);
    acc3 = __builtin_amdgcn_wmma_f32_16x16x32_bf16(false, pf, false, vf3, (short)0, acc3, false, false);
  }

  // Normalize and write y as bf16 into [B,T,C]
#pragma unroll
  for (int i = 0; i < 8; ++i) {
    const float inv = 1.0f / lrow[i];
    const int row = myQ0 + i;
    const size_t o = ((size_t)b * T_ + row) * C_ + h * D_ + nB;
    Y[o]      = f2bf(acc0[i] * inv);
    Y[o + 16] = f2bf(acc1[i] * inv);
    Y[o + 32] = f2bf(acc2[i] * inv);
    Y[o + 48] = f2bf(acc3[i] * inv);
  }
}

// ---------------------------------------------------------------------------
extern "C" void kernel_launch(void* const* d_in, const int* in_sizes, int n_in,
                              void* d_out, int out_size, void* d_ws, size_t ws_size,
                              hipStream_t stream) {
  (void)in_sizes; (void)n_in; (void)out_size; (void)ws_size;
  const float* x      = (const float*)d_in[0];  // [B,T,C]
  const float* w_attn = (const float*)d_in[1];  // [3C,C]
  const float* w_proj = (const float*)d_in[2];  // [C,C]
  float* out = (float*)d_out;                   // [B,T,C] f32

  char* ws = (char*)d_ws;                       // needs ~72 MB
  __bf16* xb   = (__bf16*)(ws);                 //  8 MB  x bf16
  __bf16* wab  = (__bf16*)(ws + 8388608);       //  6 MB  w_attn bf16
  __bf16* wpb  = (__bf16*)(ws + 14680064);      //  2 MB  w_proj bf16
  __bf16* qkvb = (__bf16*)(ws + 16777216);      // 24 MB  qkv bf16
  __bf16* qB   = (__bf16*)(ws + 41943040);      //  8 MB  Q [B,H,T,D]
  __bf16* kB   = (__bf16*)(ws + 50331648);      //  8 MB  K [B,H,T,D]
  __bf16* vB   = (__bf16*)(ws + 58720256);      //  8 MB  V [B,H,T,D]
  __bf16* yb   = (__bf16*)(ws + 67108864);      //  8 MB  y bf16

  const int nX = B_ * T_ * C_;          // 4,194,304
  const int nWA = 3 * C_ * C_;          // 3,145,728
  const int nWP = C_ * C_;              // 1,048,576
  cast_f32_bf16<<<(nX + 255) / 256, 256, 0, stream>>>(x, xb, nX);
  cast_f32_bf16<<<(nWA + 255) / 256, 256, 0, stream>>>(w_attn, wab, nWA);
  cast_f32_bf16<<<(nWP + 255) / 256, 256, 0, stream>>>(w_proj, wpb, nWP);

  // qkv = x @ w_attn^T : [4096,1024] x [3072,1024]^T
  dim3 g1(B_ * T_ / 64, 3 * C_ / 128);
  wmma_gemm_nt<true><<<g1, 128, 0, stream>>>(xb, wab, qkvb, B_ * T_, 3 * C_, C_);

  rope_split<<<(B_ * H_ * T_ * 32) / 256, 256, 0, stream>>>(qkvb, qB, kB, vB);

  dim3 g2(T_ / 64, B_ * H_);
  flash_attn_kernel<<<g2, 128, 0, stream>>>(qB, kB, vB, yb);

  // out = y @ w_proj^T : [4096,1024] x [1024,1024]^T, f32 output
  dim3 g3(B_ * T_ / 64, C_ / 128);
  wmma_gemm_nt<false><<<g3, 128, 0, stream>>>(yb, wpb, out, B_ * T_, C_, C_);
}
